// FP8ModelWrapper_82514911691348
// MI455X (gfx1250) — matrix operand order repaired and checked by reading, not compiled, not run on hardware
//
#include <hip/hip_runtime.h>

#define TOKENS  8192
#define DMODEL  4096
#define DFF     16384

typedef __bf16 v16bf __attribute__((ext_vector_type(16)));
typedef float  v8f   __attribute__((ext_vector_type(8)));

// LDS row strides (elements) for the 32-deep K tiles; padded to stay 16B
// aligned while shifting bank mapping.
#define SA 40
#define SB 40

// fp8 e4m3(fn) -> f32, branchless: place exp/mant in an f32 with the same
// field offsets, then scale by 2^(127-7). Exact for normals AND denormals.
__device__ __forceinline__ float fp8_e4m3_to_f32(unsigned int v) {
  unsigned int bits = ((v & 0x80u) << 24) | ((v & 0x7Fu) << 20);
  return __builtin_bit_cast(float, bits) * 0x1.0p+120f;
}

struct alignas(16) Frag32B { uint4 lo; uint4 hi; };

// A-fragment (16x32 bf16, ISA table): lane-half h holds K = c..c+7 (VGPR0-3)
// and K = c+16..c+23 (VGPR4-7), c = 8*h.  p points at row + 8*h.
__device__ __forceinline__ v16bf ld_frag_a(const __bf16* p) {
  Frag32B f;
  f.lo = *(const uint4*)(p);
  f.hi = *(const uint4*)(p + 16);
  return __builtin_bit_cast(v16bf, f);
}

// B-fragment (32x16 bf16, ISA table): lane-half h holds K = c..c+7 (VGPR0-3)
// and K = c+16..c+23 (VGPR4-7), c = 8*h.  p points at row + 8*h.
__device__ __forceinline__ v16bf ld_frag_b(const __bf16* p) {
  Frag32B f;
  f.lo = *(const uint4*)(p);
  f.hi = *(const uint4*)(p + 16);
  return __builtin_bit_cast(v16bf, f);
}

// Async global -> LDS copy of 16 bytes per lane (GLOBAL_LOAD_ASYNC_TO_LDS_B128,
// tracked by ASYNCcnt; no VGPR data movement).  LDS byte offset = low 32 bits
// of the generic pointer (flat LDS aperture: addr[31:0] is the LDS address).
__device__ __forceinline__ void async_copy_b128(const void* gptr, void* lptr) {
  unsigned lds_off = (unsigned)(size_t)lptr;
  asm volatile("global_load_async_to_lds_b128 %0, %1, off"
               :: "v"(lds_off), "v"(gptr) : "memory");
}

__device__ __forceinline__ void wait_asynccnt0() {
  asm volatile("s_wait_asynccnt 0x0" ::: "memory");
}

__device__ __forceinline__ float gelu_tanh(float x) {
  float x3 = x * x * x;
  float t  = tanhf(0.7978845608028654f * (x + 0.044715f * x3));
  return 0.5f * x * (1.0f + t);
}

// C = act(A[M,K](bf16) * W[K,N](fp8 e4m3) + bias[N]), C bf16.
// Block tile 128x128, K-tile 32, 256 threads = 8 waves, wave tile 64x32.
template <bool GELU>
__global__ __launch_bounds__(256) void ffn_gemm_bf16(
    const __bf16* __restrict__ A, const unsigned char* __restrict__ W,
    const __bf16* __restrict__ bias, __bf16* __restrict__ Out,
    int M, int K, int N) {
  __shared__ __bf16 ldsA[128 * SA];
  __shared__ __bf16 ldsB[128 * SB];

  const int t     = threadIdx.x;
  const int lane  = t & 31;
  const int wave  = t >> 5;
  const int wm    = wave >> 2;   // 0..1 -> 64-row half
  const int wn    = wave & 3;    // 0..3 -> 32-col quarter
  const int laneN = lane & 15;
  const int laneH = lane >> 4;   // 0/1

  const int n_base = blockIdx.x * 128;
  const int m_base = blockIdx.y * 128;

  // A staging: 128 rows x 32 cols bf16 = 512 x b128; 2 async copies/thread.
  const int arow = t >> 2;             // 0..63
  const int ac   = (t & 3) << 3;       // element offset 0/8/16/24
  // W staging: 32 rows x 128 cols fp8 = 256 x uint4; 1 per thread.
  const int krow = t >> 3;             // 0..31
  const int ncol = (t & 7) << 4;       // 0,16,...,112

  v8f acc[4][2] = {};

  const int nK = K >> 5;
  for (int kk = 0; kk < nK; ++kk) {
    const int k0 = kk << 5;

    const __bf16* gA0 = A + (size_t)(m_base + arow) * K + k0 + ac;
    const __bf16* gA1 = A + (size_t)(m_base + arow + 64) * K + k0 + ac;
    uint4 w4 = *(const uint4*)(W + (size_t)(k0 + krow) * N + n_base + ncol);

    if (kk + 1 < nK) {  // gfx1250 global_prefetch_b8 of next K tile
      __builtin_prefetch(gA0 + 32, 0, 1);
      __builtin_prefetch(W + (size_t)(k0 + 32 + krow) * N + n_base + ncol, 0, 1);
    }

    __syncthreads();  // previous iteration's LDS reads complete

    // A tile: pure byte-copy -> async DMA straight into LDS (ASYNCcnt path).
    async_copy_b128(gA0, &ldsA[arow * SA + ac]);
    async_copy_b128(gA1, &ldsA[(arow + 64) * SA + ac]);

    // W tile: fp8 -> bf16 convert + [k][n] -> [n][k] transpose via VGPRs.
    {
      unsigned int wv[4] = {w4.x, w4.y, w4.z, w4.w};
#pragma unroll
      for (int q = 0; q < 4; ++q)
#pragma unroll
        for (int b = 0; b < 4; ++b)
          ldsB[(ncol + q * 4 + b) * SB + krow] =
              (__bf16)fp8_e4m3_to_f32((wv[q] >> (8 * b)) & 0xFFu);
    }

    wait_asynccnt0();  // async A-tile landed in LDS
    __syncthreads();   // tiles visible to all waves

    v16bf af[4], bfm[2];
#pragma unroll
    for (int mt = 0; mt < 4; ++mt)
      af[mt] = ld_frag_a(&ldsA[(wm * 64 + mt * 16 + laneN) * SA + laneH * 8]);
#pragma unroll
    for (int nt = 0; nt < 2; ++nt)
      bfm[nt] = ld_frag_b(&ldsB[(wn * 32 + nt * 16 + laneN) * SB + laneH * 8]);

#pragma unroll
    for (int mt = 0; mt < 4; ++mt)
#pragma unroll
      for (int nt = 0; nt < 2; ++nt)
        acc[mt][nt] = __builtin_amdgcn_wmma_f32_16x16x32_bf16(
            false, af[mt], false, bfm[nt], (short)0, acc[mt][nt], false, false);
  }

  // Epilogue: C layout => row = r + 8*(lane>>4), col = lane&15 per 16x16 tile.
#pragma unroll
  for (int nt = 0; nt < 2; ++nt) {
    const int col  = n_base + wn * 32 + nt * 16 + laneN;
    const float bv = (float)bias[col];
#pragma unroll
    for (int mt = 0; mt < 4; ++mt) {
      const int row0 = m_base + wm * 64 + mt * 16 + laneH * 8;
#pragma unroll
      for (int r = 0; r < 8; ++r) {
        float v = acc[mt][nt][r] + bv;
        if (GELU) v = gelu_tanh(v);
        Out[(size_t)(row0 + r) * N + col] = (__bf16)v;
      }
    }
  }
}

extern "C" void kernel_launch(void* const* d_in, const int* in_sizes, int n_in,
                              void* d_out, int out_size, void* d_ws, size_t ws_size,
                              hipStream_t stream) {
  (void)in_sizes; (void)n_in; (void)out_size; (void)ws_size;
  const __bf16*         x  = (const __bf16*)d_in[0];
  const unsigned char*  w1 = (const unsigned char*)d_in[1];
  const __bf16*         b1 = (const __bf16*)d_in[2];
  const unsigned char*  w2 = (const unsigned char*)d_in[3];
  const __bf16*         b2 = (const __bf16*)d_in[4];
  __bf16*               out = (__bf16*)d_out;
  __bf16*               h   = (__bf16*)d_ws;  // 8192 x 16384 bf16 = 256 MB

  dim3 block(256);
  dim3 grid1(DFF / 128, TOKENS / 128);     // 128 x 64 blocks
  ffn_gemm_bf16<true><<<grid1, block, 0, stream>>>(x, w1, b1, h,
                                                   TOKENS, DMODEL, DFF);
  dim3 grid2(DMODEL / 128, TOKENS / 128);  // 32 x 64 blocks
  ffn_gemm_bf16<false><<<grid2, block, 0, stream>>>(h, w2, b2, out,
                                                    TOKENS, DFF, DMODEL);
}
